// GraphConv_75411035783715
// MI455X (gfx1250) — matrix operand order repaired and checked by reading, not compiled, not run on hardware
//
#include <hip/hip_runtime.h>
#include <hip/hip_bf16.h>
#include <stdint.h>

// Problem constants (from reference): O,T,DIN,H,DOUT = 50000,500000,128,256,128
static constexpr int   O_   = 50000;
static constexpr long long T_ = 500000;
static constexpr int   K1_  = 384;   // 3*DIN
static constexpr int   N1_  = 256;   // H
static constexpr int   K2_  = 256;   // H
static constexpr int   N2_  = 640;   // 2H + DOUT
static constexpr int   DOUT_= 128;

typedef __attribute__((ext_vector_type(16))) __bf16 v16bf;
typedef __attribute__((ext_vector_type(8)))  float  v8f;

union Frag16 { v16bf v; uint4 q[2]; };

static __device__ __forceinline__ v8f wmma_bf16(const Frag16& a, const Frag16& b, v8f c) {
    return __builtin_amdgcn_wmma_f32_16x16x32_bf16(false, a.v, false, b.v, (short)0, c, false, false);
}

// ---------------- utility kernels ----------------

__global__ void zero_f32(float* __restrict__ p, long long n) {
    long long i = (long long)blockIdx.x * blockDim.x + threadIdx.x;
    long long stride = (long long)gridDim.x * blockDim.x;
    for (; i < n; i += stride) p[i] = 0.0f;
}

// in: [K][N] f32 row-major  ->  out: [N][K] bf16 row-major (transposed)
__global__ void convert_wT(const float* __restrict__ in, __bf16* __restrict__ out, int K, int N) {
    int i = blockIdx.x * blockDim.x + threadIdx.x;
    if (i < K * N) {
        int k = i / N, n = i % N;
        out[(long long)n * K + k] = (__bf16)in[i];
    }
}

__global__ void count_deg(const long long* __restrict__ edges, float* __restrict__ counts) {
    long long t = (long long)blockIdx.x * blockDim.x + threadIdx.x;
    if (t < T_) {
        atomicAdd(&counts[edges[2 * t]],     1.0f);
        atomicAdd(&counts[edges[2 * t + 1]], 1.0f);
    }
}

// ---------------- edge MLP (dominant kernel) ----------------
// 32 edges per block, 256 threads = 8 waves (wave32).
// xs : [32][384] bf16 (padded stride 392)  = gathered [obj[s] | pred | obj[o]]
// h1s: [32][256] bf16 (padded stride 264)  = relu(xs @ w1a + b1a)

static constexpr int K1P = 392;  // 384 + 8 pad (784B stride, 16B aligned, bank-skewed)
static constexpr int N1P = 264;  // 256 + 8 pad

__global__ __launch_bounds__(256, 1)
void edge_mlp(const float* __restrict__ obj, const float* __restrict__ pred,
              const long long* __restrict__ edges,
              const __bf16* __restrict__ w1aT, const float* __restrict__ b1a,
              const __bf16* __restrict__ w1bT, const float* __restrict__ b1b,
              float* __restrict__ pooled, float* __restrict__ out_newp)
{
    __shared__ __bf16 xs[32][K1P];
    __shared__ __bf16 h1s[32][N1P];

    const int tid = threadIdx.x;
    const long long t0 = (long long)blockIdx.x * 32;

    // ---- gather [obj[s] | pred | obj[o]] -> bf16 LDS (4 floats per iter) ----
    for (int i = tid; i < 32 * 96; i += 256) {
        int r = i / 96;
        int c = (i % 96) * 4;
        long long t = t0 + r;
        float4 v = {0.f, 0.f, 0.f, 0.f};
        if (t < T_) {
            if (c < 128) {
                long long s = edges[2 * t];
                v = *(const float4*)(obj + s * 128 + c);
            } else if (c < 256) {
                v = *(const float4*)(pred + t * 128 + (c - 128));
            } else {
                long long o = edges[2 * t + 1];
                v = *(const float4*)(obj + o * 128 + (c - 256));
            }
        }
        xs[r][c + 0] = (__bf16)v.x; xs[r][c + 1] = (__bf16)v.y;
        xs[r][c + 2] = (__bf16)v.z; xs[r][c + 3] = (__bf16)v.w;
    }
    __syncthreads();

    const int wave = tid >> 5;
    const int lane = tid & 31;
    const int half = lane >> 4;
    const int ln16 = lane & 15;
    const int mt   = wave & 1;          // 2 M-tiles of 16 rows

    // ---- layer 1: h1 = relu(x @ w1a + b1a), N=256 (16 tiles; 4 per wave) ----
    for (int j = 0; j < 4; ++j) {
        int nt = (wave >> 1) * 4 + j;
        int n  = nt * 16 + ln16;
        v8f acc = {};
        for (int kk = 0; kk < 12; ++kk) {
            Frag16 A, B;
            const __bf16* ap = &xs[mt * 16 + ln16][kk * 32 + half * 8];
            A.q[0] = *(const uint4*)ap;
            A.q[1] = *(const uint4*)(ap + 16);
            const __bf16* bp = w1aT + (long long)n * K1_ + kk * 32 + half * 8;
            B.q[0] = *(const uint4*)bp;
            B.q[1] = *(const uint4*)(bp + 16);
            acc = wmma_bf16(A, B, acc);
        }
        float bias = b1a[n];
        #pragma unroll
        for (int r = 0; r < 8; ++r) {
            float v = acc[r] + bias;
            v = v > 0.f ? v : 0.f;
            h1s[mt * 16 + r + 8 * half][n] = (__bf16)v;
        }
    }
    __syncthreads();

    // ---- layer 2: y = relu(h1 @ w1b + b1b), N=640 (40 tiles; 10 per wave) ----
    for (int j = 0; j < 10; ++j) {
        int nt = (wave >> 1) + 4 * j;
        int n  = nt * 16 + ln16;
        v8f acc = {};
        for (int kk = 0; kk < 8; ++kk) {
            Frag16 A, B;
            const __bf16* ap = &h1s[mt * 16 + ln16][kk * 32 + half * 8];
            A.q[0] = *(const uint4*)ap;
            A.q[1] = *(const uint4*)(ap + 16);
            const __bf16* bp = w1bT + (long long)n * K2_ + kk * 32 + half * 8;
            B.q[0] = *(const uint4*)bp;
            B.q[1] = *(const uint4*)(bp + 16);
            acc = wmma_bf16(A, B, acc);
        }
        float bias = b1b[n];
        #pragma unroll
        for (int r = 0; r < 8; ++r) {
            long long t = t0 + mt * 16 + r + 8 * half;
            if (t >= T_) continue;
            float v = acc[r] + bias;
            v = v > 0.f ? v : 0.f;
            if (n < 256) {                       // new_s -> pooled[s]
                long long s = edges[2 * t];
                atomicAdd(&pooled[s * 256 + n], v);
            } else if (n < 384) {                // new_p -> output
                out_newp[t * 128 + (n - 256)] = v;
            } else {                             // new_o -> pooled[o]
                long long o = edges[2 * t + 1];
                atomicAdd(&pooled[o * 256 + (n - 384)], v);
            }
        }
    }
}

// ---------------- object MLP ----------------
// 32 objects per block; pooled/clip(count,1) -> relu(@w2a+b2a) -> relu(@w2b+b2b)

__global__ __launch_bounds__(256, 1)
void obj_mlp(const float* __restrict__ pooled, const float* __restrict__ counts,
             const __bf16* __restrict__ w2aT, const float* __restrict__ b2a,
             const __bf16* __restrict__ w2bT, const float* __restrict__ b2b,
             float* __restrict__ out_obj)
{
    __shared__ __bf16 ps[32][N1P];
    __shared__ __bf16 hs[32][N1P];

    const int tid = threadIdx.x;
    const long long m0 = (long long)blockIdx.x * 32;

    for (int i = tid; i < 32 * 64; i += 256) {
        int r = i / 64;
        int c = (i % 64) * 4;
        long long m = m0 + r;
        float4 v = {0.f, 0.f, 0.f, 0.f};
        if (m < O_) {
            v = *(const float4*)(pooled + m * 256 + c);
            float inv = 1.0f / fmaxf(counts[m], 1.0f);
            v.x *= inv; v.y *= inv; v.z *= inv; v.w *= inv;
        }
        ps[r][c + 0] = (__bf16)v.x; ps[r][c + 1] = (__bf16)v.y;
        ps[r][c + 2] = (__bf16)v.z; ps[r][c + 3] = (__bf16)v.w;
    }
    __syncthreads();

    const int wave = tid >> 5;
    const int lane = tid & 31;
    const int half = lane >> 4;
    const int ln16 = lane & 15;
    const int mt   = wave & 1;

    // layer A: N=256 (16 tiles; 4 per wave), K=256
    for (int j = 0; j < 4; ++j) {
        int nt = (wave >> 1) * 4 + j;
        int n  = nt * 16 + ln16;
        v8f acc = {};
        for (int kk = 0; kk < 8; ++kk) {
            Frag16 A, B;
            const __bf16* ap = &ps[mt * 16 + ln16][kk * 32 + half * 8];
            A.q[0] = *(const uint4*)ap;
            A.q[1] = *(const uint4*)(ap + 16);
            const __bf16* bp = w2aT + (long long)n * K2_ + kk * 32 + half * 8;
            B.q[0] = *(const uint4*)bp;
            B.q[1] = *(const uint4*)(bp + 16);
            acc = wmma_bf16(A, B, acc);
        }
        float bias = b2a[n];
        #pragma unroll
        for (int r = 0; r < 8; ++r) {
            float v = acc[r] + bias;
            v = v > 0.f ? v : 0.f;
            hs[mt * 16 + r + 8 * half][n] = (__bf16)v;
        }
    }
    __syncthreads();

    // layer B: N=128 (8 tiles; 2 per wave), K=256
    for (int j = 0; j < 2; ++j) {
        int nt = (wave >> 1) * 2 + j;
        int n  = nt * 16 + ln16;
        v8f acc = {};
        for (int kk = 0; kk < 8; ++kk) {
            Frag16 A, B;
            const __bf16* ap = &hs[mt * 16 + ln16][kk * 32 + half * 8];
            A.q[0] = *(const uint4*)ap;
            A.q[1] = *(const uint4*)(ap + 16);
            const __bf16* bp = w2bT + (long long)n * K2_ + kk * 32 + half * 8;
            B.q[0] = *(const uint4*)bp;
            B.q[1] = *(const uint4*)(bp + 16);
            acc = wmma_bf16(A, B, acc);
        }
        float bias = b2b[n];
        #pragma unroll
        for (int r = 0; r < 8; ++r) {
            long long m = m0 + mt * 16 + r + 8 * half;
            if (m >= O_) continue;
            float v = acc[r] + bias;
            v = v > 0.f ? v : 0.f;
            out_obj[m * 128 + n] = v;
        }
    }
}

// ---------------- launcher ----------------

extern "C" void kernel_launch(void* const* d_in, const int* in_sizes, int n_in,
                              void* d_out, int out_size, void* d_ws, size_t ws_size,
                              hipStream_t stream) {
    (void)in_sizes; (void)n_in; (void)out_size; (void)ws_size;

    const float*      obj   = (const float*)d_in[0];
    const float*      pred  = (const float*)d_in[1];
    const long long*  edges = (const long long*)d_in[2];
    const float*      w1a   = (const float*)d_in[3];
    const float*      b1a   = (const float*)d_in[4];
    const float*      w1b   = (const float*)d_in[5];
    const float*      b1b   = (const float*)d_in[6];
    const float*      w2a   = (const float*)d_in[7];
    const float*      b2a   = (const float*)d_in[8];
    const float*      w2b   = (const float*)d_in[9];
    const float*      b2b   = (const float*)d_in[10];

    // workspace layout (bytes, all 16B aligned)
    char* ws = (char*)d_ws;
    float*  pooled = (float*)ws;                                     // O*256 f32
    float*  counts = pooled + (long long)O_ * 256;                   // O f32
    __bf16* w1aT   = (__bf16*)(ws + (long long)51400000);            // 256x384
    __bf16* w1bT   = w1aT + (long long)N1_ * K1_;                    // 640x256
    __bf16* w2aT   = w1bT + (long long)N2_ * K2_;                    // 256x256
    __bf16* w2bT   = w2aT + (long long)K2_ * N1_;                    // 128x256

    float* out_obj  = (float*)d_out;                 // (O, 128)
    float* out_newp = out_obj + (long long)O_ * 128; // (T, 128)

    // 1) zero pooled + counts (contiguous O*257 floats)
    {
        long long n = (long long)O_ * 257;
        int blocks = 2048;
        zero_f32<<<blocks, 256, 0, stream>>>(pooled, n);
    }
    // 2) transpose-convert weights to bf16 [N][K]
    convert_wT<<<(K1_ * N1_ + 255) / 256, 256, 0, stream>>>(w1a, w1aT, K1_, N1_);
    convert_wT<<<(K2_ * N2_ + 255) / 256, 256, 0, stream>>>(w1b, w1bT, K2_, N2_);
    convert_wT<<<(K2_ * N1_ + 255) / 256, 256, 0, stream>>>(w2a, w2aT, K2_, N1_);
    convert_wT<<<(K2_ * DOUT_ + 255) / 256, 256, 0, stream>>>(w2b, w2bT, K2_, DOUT_);
    // 3) degree counts
    count_deg<<<(int)((T_ + 255) / 256), 256, 0, stream>>>(edges, counts);
    // 4) edge MLP + scatter
    {
        int blocks = (int)((T_ + 31) / 32);
        edge_mlp<<<blocks, 256, 0, stream>>>(obj, pred, edges, w1aT, b1a, w1bT, b1b,
                                             pooled, out_newp);
    }
    // 5) object MLP
    {
        int blocks = (O_ + 31) / 32;
        obj_mlp<<<blocks, 256, 0, stream>>>(pooled, counts, w2aT, b2a, w2bT, b2b, out_obj);
    }
}
